// RelationalGraphBackbone_42494406427176
// MI455X (gfx1250) — compile-verified, matrix operand order checked by reading, not run on hardware
//
#include <hip/hip_runtime.h>
#include <hip/hip_bf16.h>
#include <cstdint>
#include <cstddef>

// ---------------- problem constants ----------------
static constexpr int kNU = 50000;   // user nodes
static constexpr int kNT = 100000;  // tweet nodes
static constexpr int kD  = 256;     // hidden dim
static constexpr int kHC = 1024;    // heads * per-head channels
static constexpr int kEF = 100000, kEP = 80000, kER = 80000;
static constexpr int kL  = 3;

// ---------------- WMMA vector types ----------------
typedef __attribute__((ext_vector_type(16))) __bf16 v16bf;
typedef __attribute__((ext_vector_type(8)))  float  v8f;

__device__ __forceinline__ unsigned short f32_to_bf16(float f) {
  unsigned u = __float_as_uint(f);
  u += 0x7FFFu + ((u >> 16) & 1u);   // round to nearest even
  return (unsigned short)(u >> 16);
}

// ======================================================================
// GEMM: C[N x M] = A_bf16[N x 256] @ W[256 x M] + bias[M]
// A is row-major bf16: a lane's 16x32 WMMA A-fragment is two contiguous
// 16B runs of its row  (k = hs*8 + e  and  k = 16 + hs*8 + e), loaded
// straight from global with b128 — no LDS, no barriers.
// W is pre-swizzled (cvt_wswz_kernel) into fragment-ready order so a
// B-fragment is one contiguous 32B run per lane.
// Block = 128 threads (4 waves); wave tile = 32 rows x 64 cols
// => 8 v_wmma per K-chunk vs 12 b128 loads.
// ======================================================================
__global__ void __launch_bounds__(128)
gemm_bf16_kernel(const unsigned short* __restrict__ A,   // [N x 256] bf16 row-major
                 const unsigned short* __restrict__ Wz,  // swizzled [8][M][2][16] bf16
                 const float* __restrict__ bias,
                 float* __restrict__ C, int N, int M) {
  const int tid  = threadIdx.x;
  const int wave = tid >> 5;
  const int lane = tid & 31;
  const int hs   = lane >> 4;       // K-half select (and output row-half select)
  const int cl   = lane & 15;
  const int rowBase = blockIdx.y * 128 + wave * 32;
  const int colBlk  = blockIdx.x * 64;
  const int r0 = rowBase + cl;       // A rows for fragment 0
  const int r1 = r0 + 16;            // A rows for fragment 1

  union Frag { v16bf v; unsigned u[8]; } a0, a1, bf;
  v8f acc[8] = {};                   // [nt] rows 0-15, [nt+4] rows 16-31

  const unsigned short* arow0 = A + (size_t)r0 * kD + hs * 8;
  const unsigned short* arow1 = A + (size_t)r1 * kD + hs * 8;

#pragma unroll
  for (int kcI = 0; kcI < 8; ++kcI) {
    const int kc = kcI * 32;
    // ---- A fragments: 2x b128 each, straight from row-major global ----
    uint4 t0 = make_uint4(0, 0, 0, 0), t1 = t0, t2 = t0, t3 = t0;
    if (r0 < N) {
      t0 = ((const uint4*)(arow0 + kc))[0];
      t1 = ((const uint4*)(arow0 + kc + 16))[0];
    }
    if (r1 < N) {
      t2 = ((const uint4*)(arow1 + kc))[0];
      t3 = ((const uint4*)(arow1 + kc + 16))[0];
    }
    a0.u[0] = t0.x; a0.u[1] = t0.y; a0.u[2] = t0.z; a0.u[3] = t0.w;
    a0.u[4] = t1.x; a0.u[5] = t1.y; a0.u[6] = t1.z; a0.u[7] = t1.w;
    a1.u[0] = t2.x; a1.u[1] = t2.y; a1.u[2] = t2.z; a1.u[3] = t2.w;
    a1.u[4] = t3.x; a1.u[5] = t3.y; a1.u[6] = t3.z; a1.u[7] = t3.w;

    if (kcI < 7) {  // prefetch next K-chunk of A (gfx1250 global_prefetch)
      if (r0 < N) __builtin_prefetch(arow0 + kc + 32, 0, 3);
      if (r1 < N) __builtin_prefetch(arow1 + kc + 32, 0, 3);
    }

    // ---- 4 column sub-tiles: 1 B fragment (2x b128) feeds 2 WMMAs ----
#pragma unroll
    for (int nt = 0; nt < 4; ++nt) {
      const unsigned short* bp =
          Wz + ((size_t)(kcI * M + colBlk + nt * 16 + cl) * 2 + hs) * 16;
      uint4 b0 = ((const uint4*)bp)[0];
      uint4 b1 = ((const uint4*)bp)[1];
      bf.u[0] = b0.x; bf.u[1] = b0.y; bf.u[2] = b0.z; bf.u[3] = b0.w;
      bf.u[4] = b1.x; bf.u[5] = b1.y; bf.u[6] = b1.z; bf.u[7] = b1.w;
      acc[nt] = __builtin_amdgcn_wmma_f32_16x16x32_bf16(
          false, a0.v, false, bf.v, (short)0, acc[nt], false, false);
      acc[nt + 4] = __builtin_amdgcn_wmma_f32_16x16x32_bf16(
          false, a1.v, false, bf.v, (short)0, acc[nt + 4], false, false);
    }
  }

  // C/D layout: VGPR i -> row i (lanes 0-15) / row i+8 (lanes 16-31), col = lane&15
#pragma unroll
  for (int nt = 0; nt < 4; ++nt) {
    int gc = colBlk + nt * 16 + cl;
    float bv_ = bias[gc];
#pragma unroll
    for (int i = 0; i < 8; ++i) {
      int gr = rowBase + i + (hs << 3);
      if (gr < N) C[(size_t)gr * M + gc] = acc[nt][i] + bv_;
      int gr2 = gr + 16;
      if (gr2 < N) C[(size_t)gr2 * M + gc] = acc[nt + 4][i] + bv_;
    }
  }
}

// ======================================================================
// Convert f32 weights -> bf16 in fragment-ready (swizzled) order:
// o[(((kcI*M + n)*2 + hs)*16 + e] = bf16(W[kcI*32 + (e>>3)*16 + hs*8 + (e&7)][n])
// ======================================================================
__global__ void cvt_wswz_kernel(const float* __restrict__ W,
                                unsigned short* __restrict__ o, int M) {
  int i = blockIdx.x * blockDim.x + threadIdx.x;
  if (i >= kD * M) return;
  int e   = i & 15;
  int blk = i >> 4;
  int hs  = blk & 1;
  int nm  = blk >> 1;
  int n   = nm % M;
  int kcI = nm / M;
  int k   = kcI * 32 + ((e >> 3) << 4) + hs * 8 + (e & 7);
  o[i] = f32_to_bf16(W[(size_t)k * M + n]);
}

// ======================================================================
// LayerNorm -> bf16 normalized features. One wave per row (D=256).
// ======================================================================
__global__ void __launch_bounds__(128)
layernorm_bf16_kernel(const float* __restrict__ x, const float* __restrict__ g,
                      const float* __restrict__ b, unsigned short* __restrict__ o, int N) {
  int wave = threadIdx.x >> 5, lane = threadIdx.x & 31;
  int n = blockIdx.x * 4 + wave;
  if (n >= N) return;
  const float* xp = x + (size_t)n * kD;
  float v[8]; float s = 0.f;
#pragma unroll
  for (int j = 0; j < 8; ++j) { v[j] = xp[j * 32 + lane]; s += v[j]; }
  for (int m = 16; m; m >>= 1) s += __shfl_xor(s, m);
  float mu = s * (1.0f / 256.0f);
  float sq = 0.f;
#pragma unroll
  for (int j = 0; j < 8; ++j) { float d = v[j] - mu; sq += d * d; }
  for (int m = 16; m; m >>= 1) sq += __shfl_xor(sq, m);
  float rstd = rsqrtf(sq * (1.0f / 256.0f) + 1e-5f);
#pragma unroll
  for (int j = 0; j < 8; ++j) {
    int c = j * 32 + lane;
    o[(size_t)n * kD + c] = f32_to_bf16((v[j] - mu) * rstd * g[c] + b[c]);
  }
}

// ======================================================================
// Edge scoring + segment max (ordered-uint atomicMax). One wave per edge.
// ======================================================================
__global__ void __launch_bounds__(128)
edge_score_kernel(const int* __restrict__ src, const int* __restrict__ dst,
                  const float* __restrict__ q, const float* __restrict__ k,
                  float* __restrict__ score, unsigned* __restrict__ mbuf, int E) {
  int wave = threadIdx.x >> 5, lane = threadIdx.x & 31;
  int e = blockIdx.x * 4 + wave;
  if (e >= E) return;
  int s = src[e], d = dst[e];
  const float* qp = q + (size_t)d * kHC;
  const float* kp = k + (size_t)s * kHC;
#pragma unroll
  for (int h = 0; h < 4; ++h) {
    float a = 0.f;
#pragma unroll
    for (int j = 0; j < 8; ++j) {
      int idx = h * 256 + j * 32 + lane;
      a += qp[idx] * kp[idx];
    }
    for (int m = 16; m; m >>= 1) a += __shfl_xor(a, m);
    if (lane == 0) {
      float sc = a * 0.0625f;            // 1/sqrt(256)
      score[(size_t)e * 4 + h] = sc;
      unsigned u = __float_as_uint(sc);
      unsigned enc = (u & 0x80000000u) ? ~u : (u | 0x80000000u);
      atomicMax(mbuf + (size_t)d * 4 + h, enc);
    }
  }
}

// ======================================================================
// exp(score - max), denom += , agg[dst] += ev * v[src]. One wave per edge.
// ======================================================================
__global__ void __launch_bounds__(128)
edge_scatter_kernel(const int* __restrict__ src, const int* __restrict__ dst,
                    const float* __restrict__ score, const unsigned* __restrict__ mbuf,
                    const float* __restrict__ v, float* __restrict__ denom,
                    float* __restrict__ agg, int E) {
  int wave = threadIdx.x >> 5, lane = threadIdx.x & 31;
  int e = blockIdx.x * 4 + wave;
  if (e >= E) return;
  int s = src[e], d = dst[e];
  float ev[4];
#pragma unroll
  for (int h = 0; h < 4; ++h) {
    unsigned enc = mbuf[(size_t)d * 4 + h];
    unsigned u = (enc & 0x80000000u) ? (enc ^ 0x80000000u) : ~enc;
    ev[h] = __expf(score[(size_t)e * 4 + h] - __uint_as_float(u));
  }
  if (lane < 4) atomicAdd(denom + (size_t)d * 4 + lane, ev[lane]);
  const float* vp = v + (size_t)s * kHC;
  float* ap = agg + (size_t)d * kHC;
#pragma unroll
  for (int i = 0; i < 32; ++i) {
    int pos = i * 32 + lane;               // head index (pos>>8) is wave-uniform
    atomicAdd(ap + pos, ev[pos >> 8] * vp[pos]);
  }
}

// ======================================================================
// out[n,c] += coef * ( mean_h(agg/denom) + skip[n,c] )
// ======================================================================
__global__ void finalize_kernel(const float* __restrict__ agg, const float* __restrict__ denom,
                                const float* __restrict__ skip, float* __restrict__ out,
                                float coef, int Nd) {
  size_t idx = (size_t)blockIdx.x * blockDim.x + threadIdx.x;
  if (idx >= (size_t)Nd * kD) return;
  int n = (int)(idx >> 8), c = (int)(idx & 255);
  float t = 0.f;
#pragma unroll
  for (int h = 0; h < 4; ++h)
    t += agg[(size_t)n * kHC + h * 256 + c] / fmaxf(denom[(size_t)n * 4 + h], 1e-16f);
  out[idx] += coef * (0.25f * t + skip[idx]);
}

// ---------------- utility kernels ----------------
__global__ void zero_f32_kernel(float* p, size_t n) {
  size_t i = (size_t)blockIdx.x * blockDim.x + threadIdx.x;
  if (i < n) p[i] = 0.f;
}
__global__ void copy_f32_kernel(const float* s, float* d, size_t n) {
  size_t i = (size_t)blockIdx.x * blockDim.x + threadIdx.x;
  if (i < n) d[i] = s[i];
}
__global__ void bcast_row_kernel(const float* vec, float* d, size_t n) {
  size_t i = (size_t)blockIdx.x * blockDim.x + threadIdx.x;
  if (i < n) d[i] = vec[i & 255];
}

// ======================================================================
// host orchestration
// ======================================================================
extern "C" void kernel_launch(void* const* d_in, const int* in_sizes, int n_in,
                              void* d_out, int out_size, void* d_ws, size_t ws_size,
                              hipStream_t stream) {
  (void)in_sizes; (void)n_in; (void)out_size; (void)ws_size;

  // ---- param leaf order (jax pytree flatten: sorted dict keys) ----
  // layer l base = l*28
  //   conv rel r (0=follows,1=posts,2=rev) base = l*28 + r*8:
  //     [Wk, Wq, Ws, Wv, bk, bq, bs, bv]
  //   ln base = l*28 + 24: [tweet_b, tweet_g, user_b, user_g]
  // 84: tweet_vec  85: user_emb  86: edge_follows  87: edge_posts  88: edge_rev
  auto P = [&](int i) { return (const float*)d_in[i]; };
  const float* tweet_vec = P(84);
  const float* user_emb  = P(85);
  const int* ef = (const int*)d_in[86];
  const int* ep = (const int*)d_in[87];
  const int* er = (const int*)d_in[88];

  // ---- workspace carve-out ----
  char* wp = (char*)d_ws;
  auto alloc = [&](size_t bytes) {
    char* p = wp; wp += (bytes + 255) & ~(size_t)255; return (void*)p;
  };
  unsigned short* xun = (unsigned short*)alloc((size_t)kNU * kD * 2);
  unsigned short* xtn = (unsigned short*)alloc((size_t)kNT * kD * 2);
  unsigned short* wb  = (unsigned short*)alloc((size_t)kD * kHC * 2);
  float* qb    = (float*)alloc((size_t)kNT * kHC * 4);
  float* kb    = (float*)alloc((size_t)kNT * kHC * 4);
  float* vb    = (float*)alloc((size_t)kNT * kHC * 4);
  float* skip  = (float*)alloc((size_t)kNT * kD * 4);
  float* agg   = (float*)alloc((size_t)kNT * kHC * 4);
  float* denom = (float*)alloc((size_t)kNT * 4 * 4);
  unsigned* mbuf = (unsigned*)alloc((size_t)kNT * 4 * 4);
  float* score = (float*)alloc((size_t)kEF * 4 * 4);

  float* xu = (float*)d_out;                 // [NU, 256] residual stream
  float* xt = xu + (size_t)kNU * kD;         // [NT, 256] residual stream

  auto gemm = [&](const unsigned short* Af, const float* Wf, const float* bias,
                  float* C, int N, int M) {
    int nW = kD * M;
    cvt_wswz_kernel<<<(nW + 255) / 256, 256, 0, stream>>>(Wf, wb, M);
    dim3 grid(M / 64, (N + 127) / 128);
    gemm_bf16_kernel<<<grid, 128, 0, stream>>>(Af, wb, bias, C, N, M);
  };

  auto relation = [&](int pb, const unsigned short* xsrc, int Ns,
                      const unsigned short* xdst, int Nd,
                      const int* edges, int E, float* resid, float coef) {
    const float *Wk = P(pb + 0), *Wq = P(pb + 1), *Ws = P(pb + 2), *Wv = P(pb + 3);
    const float *bk = P(pb + 4), *bq = P(pb + 5), *bs = P(pb + 6), *bv = P(pb + 7);
    gemm(xdst, Wq, bq, qb, Nd, kHC);
    gemm(xsrc, Wk, bk, kb, Ns, kHC);
    gemm(xsrc, Wv, bv, vb, Ns, kHC);
    gemm(xdst, Ws, bs, skip, Nd, kD);
    size_t nz = (size_t)Nd * kHC;
    zero_f32_kernel<<<(unsigned)((nz + 255) / 256), 256, 0, stream>>>(agg, nz);
    zero_f32_kernel<<<(unsigned)(((size_t)Nd * 4 + 255) / 256), 256, 0, stream>>>(denom, (size_t)Nd * 4);
    zero_f32_kernel<<<(unsigned)(((size_t)Nd * 4 + 255) / 256), 256, 0, stream>>>((float*)mbuf, (size_t)Nd * 4);
    const int* srcI = edges;
    const int* dstI = edges + E;
    edge_score_kernel<<<(E + 3) / 4, 128, 0, stream>>>(srcI, dstI, qb, kb, score, mbuf, E);
    edge_scatter_kernel<<<(E + 3) / 4, 128, 0, stream>>>(srcI, dstI, score, mbuf, vb, denom, agg, E);
    size_t no = (size_t)Nd * kD;
    finalize_kernel<<<(unsigned)((no + 255) / 256), 256, 0, stream>>>(agg, denom, skip, resid, coef, Nd);
  };

  // init residual streams
  copy_f32_kernel<<<(unsigned)(((size_t)kNU * kD + 255) / 256), 256, 0, stream>>>(user_emb, xu, (size_t)kNU * kD);
  bcast_row_kernel<<<(unsigned)(((size_t)kNT * kD + 255) / 256), 256, 0, stream>>>(tweet_vec, xt, (size_t)kNT * kD);

  for (int l = 0; l < kL; ++l) {
    int lb = l * 28;
    const float* tb = P(lb + 24 + 0); const float* tg = P(lb + 24 + 1);
    const float* ub = P(lb + 24 + 2); const float* ug = P(lb + 24 + 3);
    layernorm_bf16_kernel<<<(kNU + 3) / 4, 128, 0, stream>>>(xu, ug, ub, xun, kNU);
    layernorm_bf16_kernel<<<(kNT + 3) / 4, 128, 0, stream>>>(xt, tg, tb, xtn, kNT);
    // follows: user -> user   (coef 0.5, HeteroConv mean over 2 relations)
    relation(lb + 0 * 8, xun, kNU, xun, kNU, ef, kEF, xu, 0.5f);
    // rev: tweet -> user
    relation(lb + 2 * 8, xtn, kNT, xun, kNU, er, kER, xu, 0.5f);
    // posts: user -> tweet  (single relation, coef 1.0)
    relation(lb + 1 * 8, xun, kNU, xtn, kNT, ep, kEP, xt, 1.0f);
  }
}
